// MultiHeadAttention_25065429139826
// MI455X (gfx1250) — matrix operand order repaired and checked by reading, not compiled, not run on hardware
//
#include <hip/hip_runtime.h>
#include <hip/hip_bf16.h>

typedef __bf16 bf16_t;
typedef __attribute__((ext_vector_type(16))) __bf16 v16bf;
typedef __attribute__((ext_vector_type(8)))  __bf16 v8bf;
typedef __attribute__((ext_vector_type(8)))  float  v8f;
typedef __attribute__((ext_vector_type(4)))  float  f32x4;

#define NEGV (-10000000000.0f)

// ---------------------------------------------------------------------------
// CDNA5 async copy helpers: GLOBAL_LOAD_ASYNC_TO_LDS_B128 moves 16B per lane
// from global memory directly into LDS (no VGPR round trip), tracked by
// ASYNCcnt.  LDS byte offset = value of an addrspace(3) pointer.
// ---------------------------------------------------------------------------
typedef __attribute__((address_space(3))) const void lds_cvoid_t;

__device__ __forceinline__ unsigned lds_off(const void* p) {
    return (unsigned)(size_t)(lds_cvoid_t*)p;
}

__device__ __forceinline__ void async_copy_b128(unsigned lds, const void* gptr) {
    asm volatile("global_load_async_to_lds_b128 %0, %1, off"
                 :
                 : "v"(lds), "v"((unsigned long long)(size_t)gptr)
                 : "memory");
}

__device__ __forceinline__ void wait_async0() {
    asm volatile("s_wait_asynccnt 0" ::: "memory");
}

// ---------------------------------------------------------------------------
// Kernel 1: fused QKV projection.  C[4096,1024] = X[4096,1024] @ W[1024,1024]
// + bias, output bf16.  grid.y in {0,1,2} selects q/k/v.
// Block = 128 threads (4 waves); block tile = 16(M) x 64(N), wave w owns the
// 16-wide N sub-tile.  X/W tiles staged fp32 into LDS via async-to-LDS b128
// copies, converted to bf16 in the fragment build.
// ---------------------------------------------------------------------------
__global__ __launch_bounds__(128) void proj_kernel(
    const float* __restrict__ q_in, const float* __restrict__ k_in,
    const float* __restrict__ v_in,
    const float* __restrict__ wq, const float* __restrict__ wk,
    const float* __restrict__ wv,
    const float* __restrict__ bq, const float* __restrict__ bk,
    const float* __restrict__ bv,
    bf16_t* __restrict__ Qb, bf16_t* __restrict__ Kb, bf16_t* __restrict__ Vb)
{
    const int proj = blockIdx.y;
    const float* X  = (proj == 0) ? q_in : (proj == 1) ? k_in : v_in;
    const float* W  = (proj == 0) ? wq   : (proj == 1) ? wk   : wv;
    const float* Bi = (proj == 0) ? bq   : (proj == 1) ? bk   : bv;
    bf16_t*      O  = (proj == 0) ? Qb   : (proj == 1) ? Kb   : Vb;

    const int numNt = 1024 / 64;                 // 16 N-tiles of 64
    const int m0 = (blockIdx.x / numNt) * 16;
    const int n0 = (blockIdx.x % numNt) * 64;

    __shared__ float Xs[16][32];                 // 2 KB
    __shared__ float Ws[32][64];                 // 8 KB

    const int tid  = threadIdx.x;                // 0..127
    const int lane = tid & 31;
    const int w    = tid >> 5;                   // wave id 0..3
    const int hi   = (lane >= 16) ? 1 : 0;
    const int lm   = lane & 15;

    v8f acc = {};

    for (int k0 = 0; k0 < 1024; k0 += 32) {
        // async stage: X tile 16x32 fp32 (2 KB -> one b128 per thread)
        {
            const int idx = tid * 4;
            const int r = idx >> 5, c = idx & 31;
            async_copy_b128(lds_off(&Xs[r][c]),
                            X + (size_t)(m0 + r) * 1024 + k0 + c);
        }
        // async stage: W tile 32x64 fp32 (8 KB -> four b128 per thread)
        #pragma unroll
        for (int j = 0; j < 4; ++j) {
            const int idx = tid * 4 + j * 512;
            const int r = idx >> 6, c = idx & 63;
            async_copy_b128(lds_off(&Ws[r][c]),
                            W + (size_t)(k0 + r) * 1024 + n0 + c);
        }
        if (k0 + 32 < 1024)  // global_prefetch_b8 of the next W slab
            __builtin_prefetch(W + (size_t)(k0 + 32) * 1024 + n0, 0, 1);
        wait_async0();       // drain this wave's async copies
        __syncthreads();     // publish LDS to the other waves

        // A fragment (16x32 bf16): lane m=lm holds K 0..15 (lo) / 16..31 (hi)
        v16bf a, b;
        #pragma unroll
        for (int i = 0; i < 8; ++i) {
            a[i]     = (__bf16)Xs[lm][i + (hi ? 16 : 0)];
            a[i + 8] = (__bf16)Xs[lm][i + 8 + (hi ? 16 : 0)];
        }
        // B fragment (32x16 bf16): lane n=lm holds K 0..15 (lo) / 16..31 (hi)
        #pragma unroll
        for (int i = 0; i < 16; ++i)
            b[i] = (__bf16)Ws[i + (hi ? 16 : 0)][w * 16 + lm];

        acc = __builtin_amdgcn_wmma_f32_16x16x32_bf16(
            false, a, false, b, (short)0, acc, false, false);
        __syncthreads();
    }

    const float bias = Bi[n0 + w * 16 + lm];
    #pragma unroll
    for (int r = 0; r < 8; ++r) {
        const int row = m0 + r + (hi ? 8 : 0);
        O[(size_t)row * 1024 + n0 + w * 16 + lm] = (__bf16)(acc[r] + bias);
    }
}

// ---------------------------------------------------------------------------
// Kernel 2: energy = (Q @ K^T) / 8 with mask, written fp32 to d_out.
// One wave per 16x16 energy tile; K-dim = 64 -> 2 WMMAs.  The B fragment of
// QK^T is a contiguous 32B row read of K (transpose absorbed by B layout).
// Pure bandwidth kernel: the 256 MB attention write dominates the roofline.
// ---------------------------------------------------------------------------
__global__ __launch_bounds__(128) void energy_kernel(
    const bf16_t* __restrict__ Qb, const bf16_t* __restrict__ Kb,
    const int* __restrict__ mask, float* __restrict__ attn)
{
    const int tid  = threadIdx.x;
    const int lane = tid & 31;
    const int w    = tid >> 5;
    const int hi   = (lane >= 16) ? 1 : 0;
    const int lm   = lane & 15;

    const int linear = blockIdx.x * 4 + w;       // tile id over B*H*64*64
    const int kt = linear & 63;
    const int qt = (linear >> 6) & 63;
    const int h  = (linear >> 12) & 15;
    const int b  = linear >> 16;
    const int q0 = qt * 16, c0 = kt * 16, hoff = h * 64;

    v8f acc = {};
    #pragma unroll
    for (int ks = 0; ks < 64; ks += 32) {
        v16bf a, bfrag;
        const bf16_t* qrow =
            Qb + (size_t)(b * 1024 + q0 + lm) * 1024 + hoff + ks;
        const v8bf alo = *(const v8bf*)(qrow + (hi ? 16 : 0));
        const v8bf ahi = *(const v8bf*)(qrow + 8 + (hi ? 16 : 0));
        #pragma unroll
        for (int i = 0; i < 8; ++i) { a[i] = alo[i]; a[i + 8] = ahi[i]; }

        const bf16_t* krow =
            Kb + (size_t)(b * 1024 + c0 + lm) * 1024 + hoff + ks + (hi ? 16 : 0);
        bfrag = *(const v16bf*)krow;

        acc = __builtin_amdgcn_wmma_f32_16x16x32_bf16(
            false, a, false, bfrag, (short)0, acc, false, false);
    }

    #pragma unroll
    for (int r = 0; r < 8; ++r) {
        const int qr = q0 + r + (hi ? 8 : 0);
        const int mv = mask[(size_t)b * 1024 * 1024 + (size_t)qr * 1024 + c0 + lm];
        float e = acc[r] * 0.125f;
        e = (mv == 0) ? NEGV : e;
        attn[((size_t)(b * 16 + h) * 1024 + qr) * 1024 + c0 + lm] = e;
    }
}

// ---------------------------------------------------------------------------
// Kernel 3: x = attention @ V.  K-dim = 1024 -> 32 WMMA steps.
// Block = 4 waves; block owns a 16(q) x 64(D) output tile (wave w takes the
// 16-wide D sub-tile).  V k-slab (32x64 bf16) staged in LDS via async copies.
// ---------------------------------------------------------------------------
__global__ __launch_bounds__(128) void av_kernel(
    const float* __restrict__ attn, const bf16_t* __restrict__ Vb,
    float* __restrict__ xout)
{
    __shared__ bf16_t Vs[32][64];                // 4 KB

    const int tid  = threadIdx.x;
    const int lane = tid & 31;
    const int w    = tid >> 5;
    const int hi   = (lane >= 16) ? 1 : 0;
    const int lm   = lane & 15;

    const int qt = blockIdx.x & 63;
    const int h  = (blockIdx.x >> 6) & 15;
    const int b  = blockIdx.x >> 10;
    const int q0 = qt * 16, d0 = w * 16, hoff = h * 64;

    v8f acc = {};
    for (int k0 = 0; k0 < 1024; k0 += 32) {
        // async stage of V[k0:k0+32, hoff:hoff+64] -> LDS (2 x b128 per thread)
        {
            const int idx = tid * 16;            // 16 bf16 = 32 B per thread
            const int r = idx >> 6, c = idx & 63;
            const bf16_t* g =
                Vb + (size_t)(b * 1024 + k0 + r) * 1024 + hoff + c;
            async_copy_b128(lds_off(&Vs[r][c]),     g);
            async_copy_b128(lds_off(&Vs[r][c + 8]), g + 8);
        }
        wait_async0();
        __syncthreads();

        v16bf a, bfrag;
        const float* arow =
            attn + ((size_t)(b * 16 + h) * 1024 + q0 + lm) * 1024 + k0;
        #pragma unroll
        for (int i = 0; i < 8; ++i) {
            a[i]     = (__bf16)arow[i + (hi ? 16 : 0)];
            a[i + 8] = (__bf16)arow[i + 8 + (hi ? 16 : 0)];
        }
        #pragma unroll
        for (int i = 0; i < 16; ++i)
            bfrag[i] = Vs[i + (hi ? 16 : 0)][d0 + lm];

        acc = __builtin_amdgcn_wmma_f32_16x16x32_bf16(
            false, a, false, bfrag, (short)0, acc, false, false);
        __syncthreads();
    }

    #pragma unroll
    for (int r = 0; r < 8; ++r) {
        const int qr = q0 + r + (hi ? 8 : 0);
        xout[((size_t)(b * 1024) + qr) * 1024 + hoff + d0 + lm] = acc[r];
    }
}

// ---------------------------------------------------------------------------
extern "C" void kernel_launch(void* const* d_in, const int* in_sizes, int n_in,
                              void* d_out, int out_size, void* d_ws, size_t ws_size,
                              hipStream_t stream) {
    (void)in_sizes; (void)n_in; (void)out_size; (void)ws_size;

    const float* q_in = (const float*)d_in[0];
    const float* k_in = (const float*)d_in[1];
    const float* v_in = (const float*)d_in[2];
    const int*   mask = (const int*)  d_in[3];
    const float* wq   = (const float*)d_in[4];
    const float* bq   = (const float*)d_in[5];
    const float* wk   = (const float*)d_in[6];
    const float* bk   = (const float*)d_in[7];
    const float* wv   = (const float*)d_in[8];
    const float* bv   = (const float*)d_in[9];

    // workspace: Q,K,V projections in bf16, 8 MB each
    bf16_t* Qb = (bf16_t*)d_ws;
    bf16_t* Kb = Qb + (size_t)4096 * 1024;
    bf16_t* Vb = Kb + (size_t)4096 * 1024;

    float* attn = (float*)d_out;                              // [4,16,1024,1024]
    float* xout = attn + (size_t)4 * 16 * 1024 * 1024;        // [4,1024,1024]

    // 1) QKV projections: 256 M-tiles x 16 N-tiles, y selects q/k/v
    dim3 g1(256 * 16, 3, 1);
    proj_kernel<<<g1, 128, 0, stream>>>(q_in, k_in, v_in, wq, wk, wv,
                                        bq, bk, bv, Qb, Kb, Vb);

    // 2) energy + mask: 4*16*64*64 tiles / 4 waves per block
    energy_kernel<<<65536, 128, 0, stream>>>(Qb, Kb, mask, attn);

    // 3) attention @ V: 4*16*64 blocks
    av_kernel<<<4096, 128, 0, stream>>>(attn, Vb, xout);
}
